// FuseGraphsWithAttention_31559419691726
// MI455X (gfx1250) — matrix operand order repaired and checked by reading, not compiled, not run on hardware
//
#include <hip/hip_runtime.h>

// ---------------------------------------------------------------------------
// CDNA5 (gfx1250) fused graph-attention kernel set.
// All GEMMs run through v_wmma_f32_16x16x32_bf16 (wave32, fp32 accum).
// fp32 -> bf16 conversion is done ONCE in bandwidth-bound prep kernels; the
// WMMA K-loops batch ALL 8 B-fragment loads (distinct registers) ahead of the
// 8 WMMAs so the hardware can use partial s_wait_loadcnt instead of draining.
// ---------------------------------------------------------------------------

typedef __attribute__((ext_vector_type(16))) __bf16 v16bf;
typedef __attribute__((ext_vector_type(8)))  float  v8f;

union BFrag {
    v16bf v;
    unsigned short u[16];
    uint4 q[2];
};

__device__ __forceinline__ unsigned short f2bf_u(float f) {
    union { float f; unsigned u; } a; a.f = f;
    unsigned u = a.u;
    u = (u + 0x7FFFu + ((u >> 16) & 1u)) >> 16;   // round-to-nearest-even
    return (unsigned short)u;
}

#define WMMA_BF16(A, B, C) \
    __builtin_amdgcn_wmma_f32_16x16x32_bf16(false, (A), false, (B), (short)0, (C), false, false)

// ---------------------------------------------------------------------------
// Kernel 0: bulk fp32 -> bf16 conversion (8 elements / thread, b128 stores).
// ---------------------------------------------------------------------------
__global__ __launch_bounds__(256)
void cvt_bf16_kernel(const float* __restrict__ src, unsigned short* __restrict__ dst,
                     int n8) {
    const int idx = blockIdx.x * 256 + threadIdx.x;
    if (idx >= n8) return;
    const float4* s = (const float4*)src;
    const float4 a = s[idx * 2];
    const float4 b = s[idx * 2 + 1];
    union { unsigned short u[8]; uint4 q; } o;
    o.u[0] = f2bf_u(a.x); o.u[1] = f2bf_u(a.y);
    o.u[2] = f2bf_u(a.z); o.u[3] = f2bf_u(a.w);
    o.u[4] = f2bf_u(b.x); o.u[5] = f2bf_u(b.y);
    o.u[6] = f2bf_u(b.z); o.u[7] = f2bf_u(b.w);
    ((uint4*)dst)[idx] = o.q;
}

// ---------------------------------------------------------------------------
// Kernel 1: P = relu(X @ W^T + bias), bf16 in, bf16 out [ROWS=32768, FUSE=512]
// X: [ROWS, 1024] bf16, W: [512, 1024] bf16.
// One wave: 16 rows x (8 tiles of 16 cols). Per K-step: 16 B-loads + 2 A-loads
// issued into distinct registers, then 8 WMMAs (partial-wait friendly).
// grid.x = ROWS/16 = 2048, block = 128 (wave w covers f-tiles w*8..w*8+7).
// ---------------------------------------------------------------------------
__global__ __launch_bounds__(128)
void proj_relu_kernel(const unsigned short* __restrict__ X,
                      const unsigned short* __restrict__ W,
                      const float* __restrict__ bias,
                      unsigned short* __restrict__ P) {
    const int lane = threadIdx.x & 31;
    const int wv   = threadIdx.x >> 5;
    const int rt   = blockIdx.x;            // 16-row tile
    const int hi   = lane >> 4;
    const int l15  = lane & 15;

    const unsigned short* xrow  = X + (size_t)(rt * 16 + l15) * 1024;
    const unsigned short* wbase = W + (size_t)(wv * 128 + l15) * 1024; // f-tile stride 16*1024

    v8f acc[8];
#pragma unroll
    for (int i = 0; i < 8; ++i) acc[i] = (v8f){};

    BFrag a;
    a.q[0] = *(const uint4*)(xrow + 8 * hi);
    a.q[1] = *(const uint4*)(xrow + 16 + 8 * hi);

    for (int kt = 0; kt < 32; ++kt) {       // K = 1024, step 32
        const int kb = kt * 32;
        BFrag bfr[8];
#pragma unroll
        for (int i = 0; i < 8; ++i) {
            const unsigned short* wrow = wbase + (size_t)i * 16 * 1024;
            bfr[i].q[0] = *(const uint4*)(wrow + kb + 8 * hi);
            bfr[i].q[1] = *(const uint4*)(wrow + kb + 16 + 8 * hi);
        }
        BFrag an;
        if (kt < 31) {
            an.q[0] = *(const uint4*)(xrow + kb + 32 + 8 * hi);
            an.q[1] = *(const uint4*)(xrow + kb + 48 + 8 * hi);
        }
#pragma unroll
        for (int i = 0; i < 8; ++i)
            acc[i] = WMMA_BF16(a.v, bfr[i].v, acc[i]);
        if (kt < 31) a = an;
    }

#pragma unroll
    for (int i = 0; i < 8; ++i) {
        const int f = (wv * 8 + i) * 16 + l15;
        const float bv = bias[f];
        unsigned short* orow = P + (size_t)(rt * 16 + hi * 8) * 512 + f;
#pragma unroll
        for (int r = 0; r < 8; ++r) {
            float v = acc[i][r] + bv;
            v = v > 0.f ? v : 0.f;
            orow[(size_t)r * 512] = f2bf_u(v);
        }
    }
}

// ---------------------------------------------------------------------------
// Kernel 2: VT[b][d][n] = bf16(X[b][n][d])   (per-batch transpose, LDS-tiled)
// grid = (N/32=16, D/32=32, B=64), block = 256
// ---------------------------------------------------------------------------
__global__ __launch_bounds__(256)
void transpose_bf16_kernel(const float* __restrict__ X, unsigned short* __restrict__ VT) {
    __shared__ unsigned short tile[32][33];
    const int b  = blockIdx.z;
    const int nt = blockIdx.x;
    const int dt = blockIdx.y;
    const int tx = threadIdx.x & 31;
    const int ty = threadIdx.x >> 5;        // 0..7

    const float* src = X + ((size_t)b * 512 + nt * 32) * 1024 + dt * 32;
#pragma unroll
    for (int i = 0; i < 4; ++i) {
        const int r = ty + i * 8;
        tile[r][tx] = f2bf_u(src[(size_t)r * 1024 + tx]);
    }
    __syncthreads();
    unsigned short* dst = VT + ((size_t)b * 1024 + dt * 32) * 512 + nt * 32;
#pragma unroll
    for (int i = 0; i < 4; ++i) {
        const int r = ty + i * 8;
        dst[(size_t)r * 512 + tx] = tile[tx][r];
    }
}

// ---------------------------------------------------------------------------
// Kernel 3: masked pass-through copies into the fused output.
// out[:, :, 0:1024]    = semantic_x * mask
// out[:, :, 2048:3072] = spatial_x  * mask
// grid = B*N rows, block = 256, one float4 per thread per source.
// ---------------------------------------------------------------------------
__global__ __launch_bounds__(256)
void copy_mask_kernel(const float* __restrict__ sem, const float* __restrict__ spa,
                      const int* __restrict__ counts, float* __restrict__ out) {
    const int row = blockIdx.x;
    const int b = row >> 9;
    const int n = row & 511;
    const bool valid = n < counts[b];
    const float4 z = make_float4(0.f, 0.f, 0.f, 0.f);
    const float4* s1 = (const float4*)(sem + (size_t)row * 1024);
    const float4* s2 = (const float4*)(spa + (size_t)row * 1024);
    float4* o = (float4*)(out + (size_t)row * 4096);
    o[threadIdx.x]       = valid ? s1[threadIdx.x] : z;   // cols 0..1023
    o[512 + threadIdx.x] = valid ? s2[threadIdx.x] : z;   // cols 2048..3071
}

// ---------------------------------------------------------------------------
// Kernel 4: attention pass.
//   S = Q K^T (masked over keys), row-softmax, O = softmax(S) @ V, masked rows.
// Q,K: bf16 projections [B, 512, 512]; VT: bf16 [B, 1024, 512] (V transposed).
// out: fp32 [B, 512, 4096] written at column offset colOff.
// grid = (N/16=32, B=64), block = 128 (4 waves). LDS: S 16x520 f32, P 16x520 bf16.
// ---------------------------------------------------------------------------
#define NEGV (-1e9f)

__global__ __launch_bounds__(128)
void attn_kernel(const unsigned short* __restrict__ Q,
                 const unsigned short* __restrict__ Kp,
                 const unsigned short* __restrict__ VT,
                 const int* __restrict__ counts,
                 float* __restrict__ out, int colOff) {
    __shared__ float          S[16][520];
    __shared__ unsigned short Pb[16][520];
    __shared__ float          red[16][8];
    __shared__ float          rowmax[16];
    __shared__ float          rowinv[16];

    const int b    = blockIdx.y;
    const int q0   = blockIdx.x * 16;
    const int cnt  = counts[b];
    const int lane = threadIdx.x & 31;
    const int wv   = threadIdx.x >> 5;
    const int hi   = lane >> 4;
    const int l15  = lane & 15;

    const unsigned short* qrow  = Q  + ((size_t)b * 512 + q0 + l15) * 512;
    const unsigned short* kbase = Kp + (size_t)b * 512 * 512
                                     + (size_t)(wv * 16 + l15) * 512; // key-tile stride 64*512

    // ---- Phase 1: S[16, 512] = Q-tile @ K^T, masked over keys ----
    {
        v8f acc[8];
#pragma unroll
        for (int i = 0; i < 8; ++i) acc[i] = (v8f){};

        BFrag a;
        a.q[0] = *(const uint4*)(qrow + 8 * hi);
        a.q[1] = *(const uint4*)(qrow + 16 + 8 * hi);

        for (int kt = 0; kt < 16; ++kt) {          // FUSE = 512, step 32
            const int kb = kt * 32;
            BFrag bfr[8];
#pragma unroll
            for (int i = 0; i < 8; ++i) {
                const unsigned short* krow = kbase + (size_t)i * 64 * 512;
                bfr[i].q[0] = *(const uint4*)(krow + kb + 8 * hi);
                bfr[i].q[1] = *(const uint4*)(krow + kb + 16 + 8 * hi);
            }
            BFrag an;
            if (kt < 15) {
                an.q[0] = *(const uint4*)(qrow + kb + 32 + 8 * hi);
                an.q[1] = *(const uint4*)(qrow + kb + 48 + 8 * hi);
            }
#pragma unroll
            for (int i = 0; i < 8; ++i)
                acc[i] = WMMA_BF16(a.v, bfr[i].v, acc[i]);
            if (kt < 15) a = an;
        }
#pragma unroll
        for (int i = 0; i < 8; ++i) {
            const int m = (wv + 4 * i) * 16 + l15;   // 32 key tiles over 4 waves
            const bool mvalid = m < cnt;
#pragma unroll
            for (int r = 0; r < 8; ++r)
                S[hi * 8 + r][m] = mvalid ? acc[i][r] : NEGV;
        }
    }
    __syncthreads();

    // ---- Phase 2: masked softmax per row (8 threads per row) ----
    {
        const int row = threadIdx.x >> 3;
        const int seg = threadIdx.x & 7;
        const int base = seg * 64;
        float mx = -3.4e38f;
        for (int j = 0; j < 64; ++j) mx = fmaxf(mx, S[row][base + j]);
        red[row][seg] = mx;
        __syncthreads();
        if (seg == 0) {
            float m2 = red[row][0];
#pragma unroll
            for (int j = 1; j < 8; ++j) m2 = fmaxf(m2, red[row][j]);
            rowmax[row] = m2;
        }
        __syncthreads();
        const float rm = rowmax[row];
        float sum = 0.f;
        for (int j = 0; j < 64; ++j) {
            const float e = __expf(S[row][base + j] - rm);
            sum += e;
            Pb[row][base + j] = f2bf_u(e);
        }
        red[row][seg] = sum;
        __syncthreads();
        if (seg == 0) {
            float s2 = 0.f;
#pragma unroll
            for (int j = 0; j < 8; ++j) s2 += red[row][j];
            rowinv[row] = 1.0f / s2;
        }
        __syncthreads();
    }

    // ---- Phase 3: O[16, 1024] = P @ V   (V supplied transposed: VT[d][n]) ----
    // A comes from LDS (cheap); all 8 global V fragments batched per K-step.
    const unsigned short* vbase = VT + (size_t)b * 1024 * 512
                                     + (size_t)(wv * 16 + l15) * 512; // col-tile stride 64*512
    float* obase = out + ((size_t)b * 512 + q0) * 4096 + colOff;

    for (int g = 0; g < 2; ++g) {                  // 2 groups x 8 col-tiles/wave
        v8f acc[8];
#pragma unroll
        for (int i = 0; i < 8; ++i) acc[i] = (v8f){};
        for (int kt = 0; kt < 16; ++kt) {          // keys = 512, step 32
            const int kb = kt * 32;
            BFrag bfr[8];
#pragma unroll
            for (int i = 0; i < 8; ++i) {
                const unsigned short* vrow = vbase + (size_t)(g * 8 + i) * 64 * 512;
                bfr[i].q[0] = *(const uint4*)(vrow + kb + 8 * hi);
                bfr[i].q[1] = *(const uint4*)(vrow + kb + 16 + 8 * hi);
            }
            BFrag a;
            a.q[0] = *(const uint4*)&Pb[l15][kb + 8 * hi];
            a.q[1] = *(const uint4*)&Pb[l15][kb + 16 + 8 * hi];
#pragma unroll
            for (int i = 0; i < 8; ++i)
                acc[i] = WMMA_BF16(a.v, bfr[i].v, acc[i]);
        }
#pragma unroll
        for (int i = 0; i < 8; ++i) {
            const int ct = wv + 4 * (g * 8 + i);
#pragma unroll
            for (int r = 0; r < 8; ++r) {
                const int q = hi * 8 + r;
                const bool qvalid = (q0 + q) < cnt;
                const float v = qvalid ? acc[i][r] * rowinv[q] : 0.f;
                obase[(size_t)q * 4096 + ct * 16 + l15] = v;
            }
        }
    }
}

// ---------------------------------------------------------------------------
// Host launcher
// ---------------------------------------------------------------------------
extern "C" void kernel_launch(void* const* d_in, const int* in_sizes, int n_in,
                              void* d_out, int out_size, void* d_ws, size_t ws_size,
                              hipStream_t stream) {
    (void)in_sizes; (void)n_in; (void)out_size; (void)ws_size;

    const float* sem    = (const float*)d_in[0];   // [64,512,1024]
    const float* spa    = (const float*)d_in[1];   // [64,512,1024]
    const float* Wsem   = (const float*)d_in[2];   // [512,1024]
    const float* bsem   = (const float*)d_in[3];   // [512]
    const float* Wspa   = (const float*)d_in[4];   // [512,1024]
    const float* bspa   = (const float*)d_in[5];   // [512]
    const int*   counts = (const int*)d_in[6];     // [64]
    float* out = (float*)d_out;                    // [64,512,4096]

    // Workspace layout (bytes):
    char* ws = (char*)d_ws;
    unsigned short* semX16 = (unsigned short*)(ws);                      //  64 MiB
    unsigned short* spaX16 = (unsigned short*)(ws + (size_t)67108864);   //  64 MiB
    unsigned short* Wsem16 = (unsigned short*)(ws + (size_t)134217728);  //   1 MiB
    unsigned short* Wspa16 = (unsigned short*)(ws + (size_t)135266304);  //   1 MiB
    unsigned short* semP   = (unsigned short*)(ws + (size_t)136314880);  //  32 MiB
    unsigned short* spaP   = (unsigned short*)(ws + (size_t)169869312);  //  32 MiB
    unsigned short* VTsem  = (unsigned short*)(ws + (size_t)203423744);  //  64 MiB
    unsigned short* VTspa  = (unsigned short*)(ws + (size_t)270532608);  //  64 MiB

    // One-time bf16 conversions (bandwidth-bound).
    const int nX8 = 64 * 512 * 1024 / 8;   // 4,194,304 vec8 per input tensor
    const int nW8 = 512 * 1024 / 8;        // 65,536 vec8 per weight matrix
    cvt_bf16_kernel<<<nX8 / 256, 256, 0, stream>>>(sem,  semX16, nX8);
    cvt_bf16_kernel<<<nX8 / 256, 256, 0, stream>>>(spa,  spaX16, nX8);
    cvt_bf16_kernel<<<nW8 / 256, 256, 0, stream>>>(Wsem, Wsem16, nW8);
    cvt_bf16_kernel<<<nW8 / 256, 256, 0, stream>>>(Wspa, Wspa16, nW8);

    // Projections: P = relu(X W^T + b)  (pure bf16 WMMA inner loops)
    proj_relu_kernel<<<2048, 128, 0, stream>>>(semX16, Wsem16, bsem, semP);
    proj_relu_kernel<<<2048, 128, 0, stream>>>(spaX16, Wspa16, bspa, spaP);

    // V transposes for contiguous-K WMMA B-fragments
    dim3 tg(16, 32, 64);
    transpose_bf16_kernel<<<tg, 256, 0, stream>>>(sem, VTsem);
    transpose_bf16_kernel<<<tg, 256, 0, stream>>>(spa, VTspa);

    // Masked pass-through of raw features
    copy_mask_kernel<<<32768, 256, 0, stream>>>(sem, spa, counts, out);

    // Pass A: att_sem = softmax_m(sim), weighted_spatial -> cols 1024:2048
    dim3 ag(32, 64);
    attn_kernel<<<ag, 128, 0, stream>>>(semP, spaP, VTspa, counts, out, 1024);
    // Pass B: att_spa = softmax_n(sim) == softmax over keys with Q/K swapped,
    //         weighted_semantic -> cols 3072:4096
    attn_kernel<<<ag, 128, 0, stream>>>(spaP, semP, VTsem, counts, out, 3072);
}
